// Decoder_69380901699943
// MI455X (gfx1250) — matrix operand order repaired and checked by reading, compiled, not run on hardware
//
#include <hip/hip_runtime.h>
#include <cstdint>
#include <cstddef>

// ---------------- problem constants (from reference) ----------------
#define EDIM 256      // E
#define GDIM 16       // G groups
#define NB   16       // K neighbors (S)
#define NLPT 4096     // NL low-res points
#define NHPT 16384    // NH high-res points
#define CIN_ 512
#define CSK_ 256

typedef __bf16 bf16_t;
typedef __attribute__((ext_vector_type(16))) __bf16 v16bf;
typedef __attribute__((ext_vector_type(8)))  float  v8f;

// Low 32 bits of a generic pointer to a __shared__ object == LDS byte offset
// (flat-aperture truncation rule, CDNA5 ISA 10.2).
__device__ __forceinline__ uint32_t lds_off32(const void* p) {
    return (uint32_t)(uintptr_t)p;
}
__device__ __forceinline__ void async_g2l_b128(uint32_t lds, const void* g) {
    asm volatile("global_load_async_to_lds_b128 %0, %1, off"
                 :: "v"(lds), "v"((unsigned long long)(uintptr_t)g) : "memory");
}
__device__ __forceinline__ void async_g2l_b32(uint32_t lds, const void* g) {
    asm volatile("global_load_async_to_lds_b32 %0, %1, off"
                 :: "v"(lds), "v"((unsigned long long)(uintptr_t)g) : "memory");
}

// =====================================================================
// WMMA bf16 GEMM:  C[M,N] = A[M,K] * B[K,N] (+ bias[N])
// fp32 operands in HBM. Tiles are DMA'd into LDS with
// global_load_async_to_lds_* (ASYNCcnt, double-buffered: tile k+1 in
// flight while tile k computes). B is transposed during the async copy
// via per-lane LDS scatter so both fragments read LDS contiguously;
// fp32->bf16 conversion happens in fragment assembly and co-executes
// under v_wmma_f32_16x16x32_bf16 (fp32 accumulate). wave32, 8 waves.
//   Config A: BM=64  BN=128 (2x4 waves, 2x2 16x16 tiles each)
//   Config B: BM=128 BN=16  (8x1 waves, 1x1 tile)  for N=16
// M % BM == 0, N % BN == 0, K % 32 == 0 at all call sites.
// =====================================================================
template<int BM, int BN, int WAVES_M, int WAVES_N, int WM, int WN>
__global__ __launch_bounds__(WAVES_M*WAVES_N*32)
void gemm_wmma_bf16(const float* __restrict__ A, const float* __restrict__ B,
                    const float* __restrict__ bias, float* __restrict__ C,
                    int M, int N, int Kd)
{
    constexpr int KT  = 32;
    constexpr int NT  = WAVES_M * WAVES_N * 32;
    constexpr int KP  = KT + 4;                  // fp32 pad: row stride 144B (16B mult.)
    constexpr int PA  = (BM * KT / 4) / NT;      // b128 async ops / thread / stage
    constexpr int PB  = (KT * BN) / NT;          // b32  async ops / thread / stage
    constexpr int PST = PA + PB;                 // per-wave ASYNCcnt per stage
    static_assert(PA * NT * 4 == BM * KT, "A tile must divide evenly");
    static_assert(PB * NT == KT * BN, "B tile must divide evenly");

    __shared__ __align__(16) float sAf[2][BM][KP];   // A tile, [m][k] fp32
    __shared__ __align__(16) float sBt[2][BN][KP];   // B tile, [n][k] fp32 (transposed)

    const int tid  = threadIdx.x;
    const int wave = tid >> 5;
    const int lane = tid & 31;
    const int wm   = wave / WAVES_N;
    const int wn   = wave % WAVES_N;
    const int h    = lane >> 4;      // lane half -> K sub-range
    const int l15  = lane & 15;
    const long long rowB = (long long)blockIdx.y * BM;
    const long long colB = (long long)blockIdx.x * BN;

    // Issue one stage of async tile copies (exactly PST instructions/wave).
    auto stage = [&](int buf, int k0) {
        #pragma unroll
        for (int u = 0; u < PA; ++u) {           // A: b128 chunks, row-major
            const int i  = tid + u * NT;
            const int r  = i / (KT / 4);
            const int c4 = (i % (KT / 4)) * 4;
            async_g2l_b128(lds_off32(&sAf[buf][r][c4]),
                           &A[(rowB + r) * (long long)Kd + (k0 + c4)]);
        }
        #pragma unroll
        for (int u = 0; u < PB; ++u) {           // B: b32 transpose-scatter
            const int i = tid + u * NT;
            const int r = i / BN;                // k within tile
            const int c = i % BN;                // n within tile (coalesced)
            async_g2l_b32(lds_off32(&sBt[buf][c][r]),
                          &B[(long long)(k0 + r) * N + (colB + c)]);
        }
    };

    v8f acc[WM][WN];
    #pragma unroll
    for (int i = 0; i < WM; ++i)
        #pragma unroll
        for (int j = 0; j < WN; ++j)
            acc[i][j] = (v8f){0.f,0.f,0.f,0.f,0.f,0.f,0.f,0.f};

    const int nk = Kd / KT;
    stage(0, 0);

    for (int kt = 0; kt < nk; ++kt) {
        const int buf = kt & 1;
        if (kt + 1 < nk) {
            stage(buf ^ 1, (kt + 1) * KT);
            // allow the PST just-issued ops to stay in flight; older stage done
            asm volatile("s_wait_asynccnt %0" :: "n"(PST) : "memory");
        } else {
            asm volatile("s_wait_asynccnt 0" ::: "memory");
        }
        __syncthreads();   // all waves' async data for this buf landed

        // Fragment assembly per CDNA5 wave32 WMMA layouts, cvt fp32->bf16.
        // A 16x32: lane=16h+m, elem e -> K = (e&7) + 8h + 16*(e>>3)
        v16bf afr[WM], bfr[WN];
        #pragma unroll
        for (int i = 0; i < WM; ++i) {
            const int mr = wm * WM * 16 + i * 16 + l15;
            const float* ar = &sAf[buf][mr][0];
            #pragma unroll
            for (int e = 0; e < 8; ++e)
                afr[i][e] = (bf16_t)ar[8 * h + e];
            #pragma unroll
            for (int e = 0; e < 8; ++e)
                afr[i][8 + e] = (bf16_t)ar[16 + 8 * h + e];
        }
        // B 32x16: lane=16h+n, elem e -> K = (e&7) + 8h + 16*(e>>3)
        #pragma unroll
        for (int j = 0; j < WN; ++j) {
            const int nb = wn * WN * 16 + j * 16 + l15;
            const float* bc = &sBt[buf][nb][0];
            #pragma unroll
            for (int e = 0; e < 8; ++e)
                bfr[j][e] = (bf16_t)bc[8 * h + e];
            #pragma unroll
            for (int e = 0; e < 8; ++e)
                bfr[j][8 + e] = (bf16_t)bc[16 + 8 * h + e];
        }

        #pragma unroll
        for (int i = 0; i < WM; ++i)
            #pragma unroll
            for (int j = 0; j < WN; ++j)
                acc[i][j] = __builtin_amdgcn_wmma_f32_16x16x32_bf16(
                    false, afr[i], false, bfr[j], (short)0, acc[i][j], false, false);

        __syncthreads();   // done reading buf before it is re-targeted
    }

    // C/D layout: VGPR r -> M = 8h + r ; N = l15
    #pragma unroll
    for (int i = 0; i < WM; ++i) {
        #pragma unroll
        for (int j = 0; j < WN; ++j) {
            const long long col = colB + wn * WN * 16 + j * 16 + l15;
            const float bb = bias ? bias[col] : 0.f;
            #pragma unroll
            for (int r = 0; r < 8; ++r) {
                const long long row = rowB + wm * WM * 16 + i * 16 + 8 * h + r;
                C[row * (long long)N + col] = acc[i][j][r] + bb;
            }
        }
    }
}

// ---------------- BN batch statistics: one block per channel ----------------
__global__ void colstats_k(const float* __restrict__ X, long long M, int C,
                           float* __restrict__ mean, float* __restrict__ rstd)
{
    __shared__ float s1[256], s2[256];
    const int c = blockIdx.x;
    float a = 0.f, b = 0.f;
    for (long long r = threadIdx.x; r < M; r += blockDim.x) {
        float v = X[r * C + c];
        a += v; b += v * v;
    }
    s1[threadIdx.x] = a; s2[threadIdx.x] = b;
    __syncthreads();
    for (int st = 128; st > 0; st >>= 1) {
        if ((int)threadIdx.x < st) {
            s1[threadIdx.x] += s1[threadIdx.x + st];
            s2[threadIdx.x] += s2[threadIdx.x + st];
        }
        __syncthreads();
    }
    if (threadIdx.x == 0) {
        float m = s1[0] / (float)M;
        float v = s2[0] / (float)M - m * m;
        mean[c] = m;
        rstd[c] = rsqrtf(fmaxf(v, 0.f) + 1e-5f);
    }
}

__global__ void bn_act_k(float* __restrict__ X, const float* __restrict__ mean,
                         const float* __restrict__ rstd, const float* __restrict__ g,
                         const float* __restrict__ be, long long total, int C, int relu)
{
    long long i = (long long)blockIdx.x * blockDim.x + threadIdx.x;
    if (i >= total) return;
    int c = (int)(i % C);
    float v = (X[i] - mean[c]) * rstd[c] * g[c] + be[c];
    if (relu) v = fmaxf(v, 0.f);
    X[i] = v;
}

// ---------------- gathers / elementwise glue ----------------
__global__ void gather_add_k(const float* __restrict__ src, const float* __restrict__ add,
                             const int* __restrict__ idx, float* __restrict__ dst, int Nn)
{
    long long i = (long long)blockIdx.x * blockDim.x + threadIdx.x;
    if (i >= (long long)Nn * EDIM) return;
    int n = (int)(i >> 8), e = (int)(i & 255);
    dst[i] = src[(long long)idx[n] * EDIM + e] + add[i];
}

__global__ void ref_gather_k(const int* __restrict__ lr_ref, const int* __restrict__ cluster,
                             int* __restrict__ refh, int Nn)
{
    int i = blockIdx.x * blockDim.x + threadIdx.x;
    if (i >= Nn * NB) return;
    refh[i] = lr_ref[(long long)cluster[i >> 4] * NB + (i & 15)];
}

// mode 0 (unpool): pos = coord[idx]*mask - center[n]
// mode 1 (block):  pos = (coord[idx] - center[n])*mask
__global__ void make_pos_k(const float* __restrict__ csrc, const float* __restrict__ cctr,
                           const int* __restrict__ ref, float* __restrict__ pos,
                           int Nn, int Msrc, int mode)
{
    int i = blockIdx.x * blockDim.x + threadIdx.x;
    if (i >= Nn * NB) return;
    int n = i >> 4;
    int rr = ref[i];
    float mask = (rr >= 0) ? 1.f : 0.f;
    int ix = (rr < 0) ? rr + Msrc : rr;   // jax negative-index wrap
    #pragma unroll
    for (int d = 0; d < 3; ++d) {
        float a  = csrc[(long long)ix * 3 + d];
        float c0 = cctr[(long long)n * 3 + d];
        pos[(long long)i * 3 + d] = (mode == 0) ? (a * mask - c0) : ((a - c0) * mask);
    }
}

// t = pos @ W1[3,256] + b1  (K=3: pure VALU)
__global__ void pe1_k(const float* __restrict__ pos, const float* __restrict__ W,
                      const float* __restrict__ b, float* __restrict__ T, long long rows)
{
    long long i = (long long)blockIdx.x * blockDim.x + threadIdx.x;
    if (i >= rows * EDIM) return;
    long long r = i >> 8; int c = (int)(i & 255);
    const float* p = pos + r * 3;
    T[i] = b[c] + p[0] * W[c] + p[1] * W[EDIM + c] + p[2] * W[2 * EDIM + c];
}

// rel = key[ref] - query[n] + peb
__global__ void gather_rel_k(const float* __restrict__ key, int Msrc,
                             const float* __restrict__ q, const float* __restrict__ peb,
                             const int* __restrict__ ref, float* __restrict__ rel, long long rows)
{
    long long i = (long long)blockIdx.x * blockDim.x + threadIdx.x;
    if (i >= rows * EDIM) return;
    long long r = i >> 8; int e = (int)(i & 255);
    long long n = r >> 4;
    int rr = ref[r];
    int ix = (rr < 0) ? rr + Msrc : rr;
    rel[i] = key[(long long)ix * EDIM + e] - q[n * EDIM + e] + peb[i];
}

// vg = value[ref] + peb
__global__ void gather_vadd_k(const float* __restrict__ val, int Msrc,
                              const float* __restrict__ peb, const int* __restrict__ ref,
                              float* __restrict__ vg, long long rows)
{
    long long i = (long long)blockIdx.x * blockDim.x + threadIdx.x;
    if (i >= rows * EDIM) return;
    long long r = i >> 8; int e = (int)(i & 255);
    int rr = ref[r];
    int ix = (rr < 0) ? rr + Msrc : rr;
    vg[i] = val[(long long)ix * EDIM + e] + peb[i];
}

// w = t[rows,16] @ W2[16,16] + b2
__global__ void we2_k(const float* __restrict__ t, const float* __restrict__ W2,
                      const float* __restrict__ b2, float* __restrict__ out, long long rows)
{
    long long i = (long long)blockIdx.x * blockDim.x + threadIdx.x;
    if (i >= rows * GDIM) return;
    long long r = i >> 4; int j = (int)(i & 15);
    const float* tr = t + r * GDIM;
    float a = b2[j];
    #pragma unroll
    for (int k2 = 0; k2 < 16; ++k2) a += tr[k2] * W2[k2 * GDIM + j];
    out[i] = a;
}

// softmax over the S=16 neighbor axis per (n,g), then * mask
__global__ void softmax16_k(float* __restrict__ w, const int* __restrict__ ref, int Nn)
{
    int i = blockIdx.x * blockDim.x + threadIdx.x;
    if (i >= Nn * GDIM) return;
    int n = i >> 4, g = i & 15;
    float l[NB]; float mx = -3.4e38f;
    #pragma unroll
    for (int s = 0; s < NB; ++s) {
        l[s] = w[((long long)n * NB + s) * GDIM + g];
        mx = fmaxf(mx, l[s]);
    }
    float sum = 0.f;
    #pragma unroll
    for (int s = 0; s < NB; ++s) { l[s] = __expf(l[s] - mx); sum += l[s]; }
    float inv = 1.f / sum;
    #pragma unroll
    for (int s = 0; s < NB; ++s) {
        float m = (ref[n * NB + s] >= 0) ? 1.f : 0.f;
        w[((long long)n * NB + s) * GDIM + g] = l[s] * inv * m;
    }
}

// out[n,e] = sum_s vg[n,s,e] * w[n,s,e/16]  (+ add[n,e])
__global__ void einsum_k(const float* __restrict__ vg, const float* __restrict__ w,
                         const float* __restrict__ add, float* __restrict__ out, int Nn)
{
    long long i = (long long)blockIdx.x * blockDim.x + threadIdx.x;
    if (i >= (long long)Nn * EDIM) return;
    int n = (int)(i >> 8), e = (int)(i & 255), g = e >> 4;
    float a = add ? add[i] : 0.f;
    #pragma unroll
    for (int s = 0; s < NB; ++s)
        a += vg[((long long)n * NB + s) * EDIM + e] * w[((long long)n * NB + s) * GDIM + g];
    out[i] = a;
}

__global__ void resid_relu_k(const float* __restrict__ x, const float* __restrict__ y,
                             float* __restrict__ out, long long total)
{
    long long i = (long long)blockIdx.x * blockDim.x + threadIdx.x;
    if (i >= total) return;
    out[i] = fmaxf(x[i] + y[i], 0.f);
}

// =====================================================================
// Host orchestration
// =====================================================================
struct LBR   { const float *b, *be, *g, *w; };
struct MLP2  { const float* b2; LBR l1; const float* w2; };
struct AttnP { LBR k; MLP2 pe; LBR q; const float *vb, *vw; MLP2 we; };
struct BlockP{ AttnP attn; const float *fc1, *fc3, *n1b, *n1g, *n2b, *n2g, *n3b, *n3g; };
struct UpP   { LBR key; MLP2 pe; LBR proj; LBR proj_skip; LBR query; MLP2 we; };

extern "C" void kernel_launch(void* const* d_in, const int* in_sizes, int n_in,
                              void* d_out, int out_size, void* d_ws, size_t ws_size,
                              hipStream_t stream)
{
    (void)in_sizes; (void)out_size; (void)ws_size;
    if (n_in < 95) return;   // 7 tensors + 2*30 block leaves + 28 up leaves

    const float* coord      = (const float*)d_in[0];
    const float* feat       = (const float*)d_in[1];
    const float* skip_coord = (const float*)d_in[2];
    const float* skip_feat  = (const float*)d_in[3];
    const int*   cluster    = (const int*)  d_in[4];
    const int*   hr_ref     = (const int*)  d_in[6];

    // jax pytree flatten order: dict keys sorted; 'blocks' < 'up'.
    int p = 7;
    auto F = [&]() { return (const float*)d_in[p++]; };
    auto rdLBR  = [&]() { LBR l; l.b = F(); l.be = F(); l.g = F(); l.w = F(); return l; };
    auto rdMLP2 = [&]() { MLP2 m; m.b2 = F(); m.l1 = rdLBR(); m.w2 = F(); return m; };

    BlockP blk[2];
    for (int i = 0; i < 2; ++i) {
        BlockP B;
        B.attn.k  = rdLBR();
        B.attn.pe = rdMLP2();
        B.attn.q  = rdLBR();
        B.attn.vb = F(); B.attn.vw = F();
        B.attn.we = rdMLP2();
        B.fc1 = F(); B.fc3 = F();
        B.n1b = F(); B.n1g = F(); B.n2b = F(); B.n2g = F(); B.n3b = F(); B.n3g = F();
        blk[i] = B;
    }
    UpP up;
    up.key = rdLBR(); up.pe = rdMLP2(); up.proj = rdLBR();
    up.proj_skip = rdLBR(); up.query = rdLBR(); up.we = rdMLP2();

    // ---------------- workspace bump allocator ----------------
    char* base = (char*)d_ws;
    size_t off = 0;
    auto allocf = [&](size_t nelem) -> float* {
        float* ptr = (float*)(base + off);
        off = (off + nelem * sizeof(float) + 255) & ~(size_t)255;
        return ptr;
    };
    const size_t NE  = (size_t)NHPT * EDIM;           // 4.19M
    const size_t BIG = (size_t)NHPT * NB * EDIM;      // 67.1M
    float* sk    = allocf(NE);
    float* lr    = allocf((size_t)NLPT * EDIM);
    float* hr    = allocf(NE);
    float* keyb  = allocf(NE);
    float* qryb  = allocf(NE);
    float* vbuf  = allocf(NE);
    float* x     = allocf(NE);
    float* x1    = allocf(NE);
    float* aout  = allocf(NE);
    int*   refh  = (int*)allocf((size_t)NHPT * NB);
    float* pos   = allocf((size_t)NHPT * NB * 3);
    float* wt    = allocf((size_t)NHPT * NB * GDIM);
    float* wsm   = allocf((size_t)NHPT * NB * GDIM);
    float* meanb = allocf(512);
    float* rstdb = allocf(512);
    float* t1    = allocf(BIG);     // pe-l1 activations, then rel
    float* peb   = allocf(BIG);
    float* vgb   = allocf(BIG);

    // ---------------- launch helpers ----------------
    auto gemm256 = [&](const float* A, const float* W, const float* b, float* Cm,
                       int M, int N, int Kd) {
        dim3 g(N / 128, M / 64);
        gemm_wmma_bf16<64, 128, 2, 4, 2, 2><<<g, 256, 0, stream>>>(A, W, b, Cm, M, N, Kd);
    };
    auto gemm16 = [&](const float* A, const float* W, const float* b, float* Cm,
                      int M, int Kd) {
        dim3 g(1, M / 128);
        gemm_wmma_bf16<128, 16, 8, 1, 1, 1><<<g, 256, 0, stream>>>(A, W, b, Cm, M, 16, Kd);
    };
    auto bnstats = [&](float* X, long long M, int C) {
        colstats_k<<<C, 256, 0, stream>>>(X, M, C, meanb, rstdb);
    };
    auto bnact = [&](float* X, const float* g_, const float* b_, long long M, int C, int relu) {
        long long tot = M * C;
        bn_act_k<<<(unsigned)((tot + 255) / 256), 256, 0, stream>>>(X, meanb, rstdb, g_, b_, tot, C, relu);
    };
    auto lbr_apply = [&](const float* A, const LBR& P, float* Cm, int M, int Kd) {
        gemm256(A, P.w, P.b, Cm, M, EDIM, Kd);
        bnstats(Cm, M, EDIM);
        bnact(Cm, P.g, P.be, M, EDIM, 1);
    };
    const long long Mr = (long long)NHPT * NB;   // 262144 rows
    auto pe_path = [&](const MLP2& pe) {
        pe1_k<<<(unsigned)((Mr * EDIM + 255) / 256), 256, 0, stream>>>(pos, pe.l1.w, pe.l1.b, t1, Mr);
        bnstats(t1, Mr, EDIM);
        bnact(t1, pe.l1.g, pe.l1.be, Mr, EDIM, 1);
        gemm256(t1, pe.w2, pe.b2, peb, (int)Mr, EDIM, EDIM);
    };
    auto we_path = [&](const MLP2& we, const int* ref) {
        gemm16(t1 /*rel*/, we.l1.w, we.l1.b, wt, (int)Mr, EDIM);
        bnstats(wt, Mr, GDIM);
        bnact(wt, we.l1.g, we.l1.be, Mr, GDIM, 1);
        we2_k<<<(unsigned)((Mr * GDIM + 255) / 256), 256, 0, stream>>>(wt, we.w2, we.b2, wsm, Mr);
        softmax16_k<<<(NHPT * GDIM + 255) / 256, 256, 0, stream>>>(wsm, ref, NHPT);
    };
    auto attn_core = [&](const float* keysrc, int Msrc, const float* query,
                         const float* valuesrc, const MLP2& pe, const MLP2& we,
                         const int* ref, const float* addsrc, float* outp) {
        pe_path(pe);
        gather_rel_k<<<(unsigned)((Mr * EDIM + 255) / 256), 256, 0, stream>>>(
            keysrc, Msrc, query, peb, ref, t1, Mr);
        gather_vadd_k<<<(unsigned)((Mr * EDIM + 255) / 256), 256, 0, stream>>>(
            valuesrc, Msrc, peb, ref, vgb, Mr);
        we_path(we, ref);
        einsum_k<<<(unsigned)(((long long)NHPT * EDIM + 255) / 256), 256, 0, stream>>>(
            vgb, wsm, addsrc, outp, NHPT);
    };

    // ==================== _unpool_ca ====================
    gemm256(skip_feat, up.proj_skip.w, up.proj_skip.b, sk, NHPT, EDIM, CSK_);
    bnstats(sk, NHPT, EDIM); bnact(sk, up.proj_skip.g, up.proj_skip.be, NHPT, EDIM, 1);

    gemm256(feat, up.proj.w, up.proj.b, lr, NLPT, EDIM, CIN_);
    bnstats(lr, NLPT, EDIM); bnact(lr, up.proj.g, up.proj.be, NLPT, EDIM, 1);

    gather_add_k<<<(unsigned)((NE + 255) / 256), 256, 0, stream>>>(lr, sk, cluster, hr, NHPT);

    lbr_apply(lr, up.key, keyb, NLPT, EDIM);
    lbr_apply(hr, up.query, qryb, NHPT, EDIM);

    ref_gather_k<<<(NHPT * NB + 255) / 256, 256, 0, stream>>>((const int*)d_in[5], cluster, refh, NHPT);
    make_pos_k<<<(NHPT * NB + 255) / 256, 256, 0, stream>>>(coord, skip_coord, refh, pos,
                                                            NHPT, NLPT, /*mode=*/0);
    attn_core(keyb, NLPT, qryb, lr, up.pe, up.we, refh, /*add=*/sk, x);

    // ==================== blocks ====================
    for (int b = 0; b < 2; ++b) {
        const BlockP& B = blk[b];
        float* finaldst = (b == 1) ? (float*)d_out : x;

        gemm256(x, B.fc1, nullptr, x1, NHPT, EDIM, EDIM);
        bnstats(x1, NHPT, EDIM); bnact(x1, B.n1g, B.n1b, NHPT, EDIM, 1);

        lbr_apply(x1, B.attn.q, qryb, NHPT, EDIM);
        lbr_apply(x1, B.attn.k, keyb, NHPT, EDIM);
        gemm256(x1, B.attn.vw, B.attn.vb, vbuf, NHPT, EDIM, EDIM);

        make_pos_k<<<(NHPT * NB + 255) / 256, 256, 0, stream>>>(skip_coord, skip_coord, hr_ref,
                                                                pos, NHPT, NHPT, /*mode=*/1);
        attn_core(keyb, NHPT, qryb, vbuf, B.attn.pe, B.attn.we, hr_ref, nullptr, aout);

        bnstats(aout, NHPT, EDIM); bnact(aout, B.n2g, B.n2b, NHPT, EDIM, 1);

        gemm256(aout, B.fc3, nullptr, x1, NHPT, EDIM, EDIM);
        bnstats(x1, NHPT, EDIM); bnact(x1, B.n3g, B.n3b, NHPT, EDIM, 0);

        resid_relu_k<<<(unsigned)((NE + 255) / 256), 256, 0, stream>>>(x, x1, finaldst, (long long)NE);
    }
}
